// VSSBlock_46978352284130
// MI455X (gfx1250) — matrix operand order repaired and checked by reading, not compiled, not run on hardware
//
#include <hip/hip_runtime.h>
#include <hip/hip_bf16.h>
#include <math.h>

#if __has_include(<hip/amd_detail/amd_gfx1250_TDM.h>)
#define TDM_6ARG 1
#else
#define TDM_6ARG 0
#endif

#define Bb 2
#define Hh 48
#define Ww 48
#define Cc 192
#define Dd 384
#define Kk 4
#define Rr 12
#define Nn 16
#define Ll (Hh*Ww)      // 2304
#define BL (Bb*Ll)      // 4608
#define CP 48           // R+2N=44 padded to 48
#define E2 (2*Dd)       // 768

typedef __hip_bfloat16 bf16;
typedef __attribute__((ext_vector_type(16))) __bf16 v16bf;
typedef __attribute__((ext_vector_type(8)))  float  v8f;
typedef __attribute__((ext_vector_type(4))) unsigned int u32x4;
typedef __attribute__((ext_vector_type(8))) int          i32x8;
typedef __attribute__((ext_vector_type(4))) int          i32x4;

union AU { v16bf v; uint4 u[2]; };

// ---- WMMA tile loaders (16-bit A 16x32 / B 32x16 striping, wave32) ----
__device__ inline void load_a(AU& a, const bf16* rowbase_k0, int half) {
    // lane half=0 holds K 0..7 and 16..23 of its row; half=1 holds 8..15 and 24..31
    a.u[0] = *(const uint4*)(rowbase_k0 + half * 8);
    a.u[1] = *(const uint4*)(rowbase_k0 + 16 + half * 8);
}
__device__ inline void load_b(AU& b, const bf16* colbase_k0, int half) {
    // lane = column; half=0 holds K 0..7 and 16..23, half=1 holds K 8..15 and 24..31 (W[n,k])
    const bf16* p = colbase_k0 + half * 8;
    b.u[0] = *(const uint4*)(p);
    b.u[1] = *(const uint4*)(p + 16);
}

// scan-direction row permutation: physical flat (h*W+w) index for scan position r
__device__ inline int perm_l(int k, int r) {
    switch (k) {
        case 0:  return r;
        case 1:  return (r % Hh) * Ww + (r / Hh);
        case 2:  return Ll - 1 - r;
        default: { int rr = Ll - 1 - r; return (rr % Hh) * Ww + (rr / Hh); }
    }
}

__device__ inline float sigmoidf_(float x) { return 1.0f / (1.0f + expf(-x)); }

// ---------------- kernel 1: LayerNorm over C, emit bf16 ----------------
__global__ void k_ln(const float* __restrict__ x, const float* __restrict__ g,
                     const float* __restrict__ b, bf16* __restrict__ xn) {
    __shared__ float ss[256], sq[256];
    int row = blockIdx.x, t = threadIdx.x;
    float v = (t < Cc) ? x[(size_t)row * Cc + t] : 0.f;
    ss[t] = v; sq[t] = v * v;
    __syncthreads();
    for (int s = 128; s > 0; s >>= 1) {
        if (t < s) { ss[t] += ss[t + s]; sq[t] += sq[t + s]; }
        __syncthreads();
    }
    float m   = ss[0] / (float)Cc;
    float var = sq[0] / (float)Cc - m * m;
    float inv = rsqrtf(var + 1e-5f);
    if (t < Cc) xn[(size_t)row * Cc + t] = __float2bfloat16((v - m) * inv * g[t] + b[t]);
}

// ---------------- weight converts ----------------
__global__ void k_cvt(const float* __restrict__ src, bf16* __restrict__ dst, int n) {
    int i = blockIdx.x * blockDim.x + threadIdx.x;
    if (i < n) dst[i] = __float2bfloat16(src[i]);
}
__global__ void k_cvt_xproj(const float* __restrict__ src, bf16* __restrict__ dst) {
    int i = blockIdx.x * blockDim.x + threadIdx.x;
    if (i >= Kk * CP * Dd) return;
    int d = i % Dd, c = (i / Dd) % CP, k = i / (Dd * CP);
    float v = (c < Rr + 2 * Nn) ? src[((size_t)k * (Rr + 2 * Nn) + c) * Dd + d] : 0.f;
    dst[i] = __float2bfloat16(v);
}

// ---------------- kernel: xz = xn @ in_w^T  (M=4608,N=768,K=192) ----------------
__global__ void k_gemm_xz(const bf16* __restrict__ xn, const bf16* __restrict__ w,
                          float* __restrict__ xz) {
    int wave = (blockIdx.x * blockDim.x + threadIdx.x) >> 5;
    int lane = threadIdx.x & 31;
    const int MT = BL / 16;                 // 288
    int mt = wave % MT, nt = wave / MT;     // nt < 48
    int r = lane & 15, half = lane >> 4;
    const bf16* arow = xn + (size_t)(mt * 16 + r) * Cc;
    const bf16* brow = w  + (size_t)(nt * 16 + r) * Cc;
    v8f acc = {};
    for (int k0 = 0; k0 < Cc; k0 += 32) {
        __builtin_prefetch(arow + k0 + 32, 0, 1);
        AU a, b;
        load_a(a, arow + k0, half);
        load_b(b, brow + k0, half);
        acc = __builtin_amdgcn_wmma_f32_16x16x32_bf16(false, a.v, false, b.v,
                                                      (short)0, acc, false, false);
    }
    float* crow = xz + (size_t)(mt * 16 + half * 8) * E2 + nt * 16 + r;
    #pragma unroll
    for (int v = 0; v < 8; ++v) crow[(size_t)v * E2] = acc[v];
}

// ---------------- depthwise 3x3 conv + bias + SiLU ----------------
__global__ void k_conv(const float* __restrict__ xz, const float* __restrict__ cw,
                       const float* __restrict__ cb, float* __restrict__ xcf,
                       bf16* __restrict__ xcb) {
    size_t i = (size_t)blockIdx.x * blockDim.x + threadIdx.x;
    if (i >= (size_t)Bb * Ll * Dd) return;
    int d = i % Dd;
    int l = (int)((i / Dd) % Ll);
    int b = (int)(i / ((size_t)Dd * Ll));
    int h = l / Ww, w = l % Ww;
    float acc = cb[d];
    #pragma unroll
    for (int dh = -1; dh <= 1; ++dh)
        #pragma unroll
        for (int dw = -1; dw <= 1; ++dw) {
            int hh = h + dh, ww2 = w + dw;
            if (hh >= 0 && hh < Hh && ww2 >= 0 && ww2 < Ww)
                acc += xz[((size_t)b * Ll + hh * Ww + ww2) * E2 + d] *
                       cw[d * 9 + (dh + 1) * 3 + (dw + 1)];
        }
    float s = acc * sigmoidf_(acc);
    xcf[i] = s;
    xcb[i] = __float2bfloat16(s);
}

// -------- x_dbl GEMM with TDM-staged weights in LDS --------
// Each block handles 8 tiles of one (b, kdir); the 48x384 bf16 projection
// matrix (36 KB) is DMA'd into LDS once per workgroup by the Tensor Data
// Mover, completion enforced with s_wait_tensorcnt, then all 8 waves serve
// their WMMA B-operands from LDS.
#define XDBL_TILES   (3 * (Ll / 16))        // 432 tiles per (b,kdir)
#define XDBL_TPB     (XDBL_TILES / 8)       // 54 blocks per (b,kdir)
__global__ void k_gemm_xdbl(const bf16* __restrict__ xc, const bf16* __restrict__ wp,
                            float* __restrict__ xdbl) {
    __shared__ bf16 wlds[CP * Dd];          // 36,864 bytes
    int tg = blockIdx.x % XDBL_TPB;
    int kd = (blockIdx.x / XDBL_TPB) % Kk;
    int b  =  blockIdx.x / (XDBL_TPB * Kk);

    // --- TDM: stage wp[kd] (48 rows x 384 bf16) into LDS ---
    if (threadIdx.x < 32) {
        unsigned long long ga = (unsigned long long)(size_t)(wp + (size_t)kd * CP * Dd);
        unsigned lds_off = (unsigned)(size_t)&wlds[0];
        u32x4 g0 = {};
        g0.x = 1u;                                   // count=1, is_restore=0, gather off
        g0.y = lds_off;                              // lds_addr [63:32]
        g0.z = (unsigned)(ga & 0xFFFFFFFFu);         // global_addr lo
        g0.w = (unsigned)((ga >> 32) & 0x01FFFFFFu)  // global_addr [56:32]
             | (2u << 30);                           // type=2 ("image")
        i32x8 g1 = {};
        g1[0] = (1 << 16);                           // workgroup_mask=0, data_size=1 (2 bytes)
        g1[1] = (Dd & 0xFFFF) << 16;                 // tensor_dim0[15:0] in bits 63:48
        g1[2] = ((Dd >> 16) & 0xFFFF)                // tensor_dim0[31:16]
              | ((CP & 0xFFFF) << 16);               // tensor_dim1[15:0]
        g1[3] = ((CP >> 16) & 0xFFFF)                // tensor_dim1[31:16]
              | ((Dd & 0xFFFF) << 16);               // tile_dim0 = 384
        g1[4] = CP;                                  // tile_dim1 = 48, tile_dim2 = 0
        g1[5] = Dd;                                  // tensor_dim0_stride[31:0] = 384
        g1[6] = ((Dd * CP) & 0xFFFF) << 16;          // dim0_stride[47:32]=0, dim1_stride[15:0]
        g1[7] = (Dd * CP) >> 16;                     // tensor_dim1_stride[47:16]
        i32x4 g2 = {};                               // 2-D tensor: groups 2/3 unused
        i32x4 g3 = {};
#if TDM_6ARG
        i32x8 g4 = {};
        __builtin_amdgcn_tensor_load_to_lds(g0, g1, g2, g3, g4, 0);
#else
        __builtin_amdgcn_tensor_load_to_lds(g0, g1, g2, g3, 0);
#endif
        __builtin_amdgcn_s_wait_tensorcnt(0);
    }
    __syncthreads();

    int wave = threadIdx.x >> 5;
    int lane = threadIdx.x & 31;
    int tile = tg * 8 + wave;                // 0..431 within (b,kd)
    int mt = tile / 3, nt = tile % 3;
    int r = lane & 15, half = lane >> 4;
    int pl = perm_l(kd, mt * 16 + r);
    const bf16* arow = xc + ((size_t)b * Ll + pl) * Dd;
    const bf16* brow = &wlds[(nt * 16 + r) * Dd];    // B served from LDS
    v8f acc = {};
    for (int k0 = 0; k0 < Dd; k0 += 32) {
        __builtin_prefetch(arow + k0 + 32, 0, 1);
        AU a, bm;
        load_a(a, arow + k0, half);
        load_b(bm, brow + k0, half);
        acc = __builtin_amdgcn_wmma_f32_16x16x32_bf16(false, a.v, false, bm.v,
                                                      (short)0, acc, false, false);
    }
    float* base = xdbl + ((size_t)(b * Kk + kd) * Ll + mt * 16 + half * 8) * CP + nt * 16 + r;
    #pragma unroll
    for (int v = 0; v < 8; ++v) base[(size_t)v * CP] = acc[v];
}

// ---------------- selective scan: one block per (b,k), thread = channel d ----------------
__global__ void k_scan(const float* __restrict__ xcf, const float* __restrict__ xdbl,
                       const float* __restrict__ dt_w, const float* __restrict__ dt_b,
                       const float* __restrict__ A_logs, const float* __restrict__ Ds,
                       float* __restrict__ ys) {
    int b  = blockIdx.x / Kk;
    int kd = blockIdx.x % Kk;
    int d  = threadIdx.x;           // 0..383
    __shared__ float sbc[CP];       // [0..11]=dts, [12..27]=B, [28..43]=C
    float dtw[Rr];
    #pragma unroll
    for (int r = 0; r < Rr; ++r) dtw[r] = dt_w[((size_t)kd * Dd + d) * Rr + r];
    float dtb = dt_b[kd * Dd + d];
    float Dsv = Ds[kd * Dd + d];
    float Ar[Nn], h[Nn];
    #pragma unroll
    for (int n = 0; n < Nn; ++n) {
        Ar[n] = -expf(A_logs[((size_t)kd * Dd + d) * Nn + n]);
        h[n]  = 0.f;
    }
    const float* xd = xdbl + (size_t)(b * Kk + kd) * Ll * CP;
    const float* uc = xcf  + (size_t)b * Ll * Dd;
    float*       yo = ys   + (size_t)(b * Kk + kd) * Ll * Dd;
    for (int l = 0; l < Ll; ++l) {
        if (d < Rr + 2 * Nn) sbc[d] = xd[(size_t)l * CP + d];
        __syncthreads();
        float pre = dtb;
        #pragma unroll
        for (int r = 0; r < Rr; ++r) pre += sbc[r] * dtw[r];
        float delta = (pre > 20.f) ? pre : log1pf(expf(pre));   // softplus
        float u  = uc[(size_t)perm_l(kd, l) * Dd + d];
        float du = delta * u;
        float y = 0.f;
        #pragma unroll
        for (int n = 0; n < Nn; ++n) {
            float dA = expf(delta * Ar[n]);
            h[n] = dA * h[n] + du * sbc[12 + n];
            y   += h[n] * sbc[28 + n];
        }
        yo[(size_t)l * Dd + d] = y + Dsv * u;
        __syncthreads();
    }
}

// ---------------- merge the 4 scan directions back to canonical (b,l,d) ----------------
__global__ void k_combine(const float* __restrict__ ys, float* __restrict__ yc) {
    size_t i = (size_t)blockIdx.x * blockDim.x + threadIdx.x;
    if (i >= (size_t)Bb * Ll * Dd) return;
    int d = i % Dd;
    int l = (int)((i / Dd) % Ll);
    int b = (int)(i / ((size_t)Dd * Ll));
    int h = l / Ww, w = l % Ww;
    int lt = w * Hh + h;
    const float* base = ys + (size_t)b * Kk * Ll * Dd;
    float s = base[((size_t)0 * Ll + l)            * Dd + d]
            + base[((size_t)2 * Ll + (Ll - 1 - l))  * Dd + d]
            + base[((size_t)1 * Ll + lt)            * Dd + d]
            + base[((size_t)3 * Ll + (Ll - 1 - lt)) * Dd + d];
    yc[i] = s;
}

// ---------------- output LN over D + SiLU(z) gate, emit bf16 ----------------
__global__ void k_ln_gate(const float* __restrict__ y, const float* __restrict__ xz,
                          const float* __restrict__ g, const float* __restrict__ b,
                          bf16* __restrict__ yg) {
    __shared__ float ss[512], sq[512];
    int row = blockIdx.x, t = threadIdx.x;
    float v = (t < Dd) ? y[(size_t)row * Dd + t] : 0.f;
    ss[t] = v; sq[t] = v * v;
    __syncthreads();
    for (int s = 256; s > 0; s >>= 1) {
        if (t < s) { ss[t] += ss[t + s]; sq[t] += sq[t + s]; }
        __syncthreads();
    }
    float m   = ss[0] / (float)Dd;
    float var = sq[0] / (float)Dd - m * m;
    float inv = rsqrtf(var + 1e-5f);
    if (t < Dd) {
        float z = xz[(size_t)row * E2 + Dd + t];
        float gate = z * sigmoidf_(z);
        yg[(size_t)row * Dd + t] = __float2bfloat16(((v - m) * inv * g[t] + b[t]) * gate);
    }
}

// ---------------- out GEMM + residual: out = x + yg @ out_w^T (M=4608,N=192,K=384) ----------------
__global__ void k_gemm_out(const bf16* __restrict__ yg, const bf16* __restrict__ w,
                           const float* __restrict__ x, float* __restrict__ out) {
    int wave = (blockIdx.x * blockDim.x + threadIdx.x) >> 5;
    int lane = threadIdx.x & 31;
    int nt = wave % (Cc / 16);   // 12
    int mt = wave / (Cc / 16);   // 288
    int r = lane & 15, half = lane >> 4;
    const bf16* arow = yg + (size_t)(mt * 16 + r) * Dd;
    const bf16* brow = w  + (size_t)(nt * 16 + r) * Dd;
    v8f acc = {};
    for (int k0 = 0; k0 < Dd; k0 += 32) {
        __builtin_prefetch(arow + k0 + 32, 0, 1);
        AU a, bm;
        load_a(a, arow + k0, half);
        load_b(bm, brow + k0, half);
        acc = __builtin_amdgcn_wmma_f32_16x16x32_bf16(false, a.v, false, bm.v,
                                                      (short)0, acc, false, false);
    }
    int row = mt * 16 + half * 8, col = nt * 16 + r;
    #pragma unroll
    for (int v = 0; v < 8; ++v)
        out[(size_t)(row + v) * Cc + col] = acc[v] + x[(size_t)(row + v) * Cc + col];
}

extern "C" void kernel_launch(void* const* d_in, const int* in_sizes, int n_in,
                              void* d_out, int out_size, void* d_ws, size_t ws_size,
                              hipStream_t stream) {
    const float* x       = (const float*)d_in[0];
    const float* ln_g    = (const float*)d_in[1];
    const float* ln_b    = (const float*)d_in[2];
    const float* in_w    = (const float*)d_in[3];
    const float* conv_w  = (const float*)d_in[4];
    const float* conv_b  = (const float*)d_in[5];
    const float* xproj_w = (const float*)d_in[6];
    const float* dt_w    = (const float*)d_in[7];
    const float* dt_b    = (const float*)d_in[8];
    const float* A_logs  = (const float*)d_in[9];
    const float* Ds      = (const float*)d_in[10];
    const float* on_g    = (const float*)d_in[11];
    const float* on_b    = (const float*)d_in[12];
    const float* out_w   = (const float*)d_in[13];
    float* out = (float*)d_out;

    char* ws = (char*)d_ws;
    size_t off = 0;
    auto carve = [&](size_t bytes) { size_t r = off; off += (bytes + 255) & ~(size_t)255; return r; };
    bf16*  xn_bf   = (bf16*)(ws + carve((size_t)BL * Cc * 2));
    bf16*  w_in_bf = (bf16*)(ws + carve((size_t)E2 * Cc * 2));
    bf16*  w_xp_bf = (bf16*)(ws + carve((size_t)Kk * CP * Dd * 2));
    bf16*  w_o_bf  = (bf16*)(ws + carve((size_t)Cc * Dd * 2));
    float* xz      = (float*)(ws + carve((size_t)BL * E2 * 4));
    float* xc_f    = (float*)(ws + carve((size_t)BL * Dd * 4));
    bf16*  xc_bf   = (bf16*)(ws + carve((size_t)BL * Dd * 2));
    float* xdbl    = (float*)(ws + carve((size_t)Bb * Kk * Ll * CP * 4));
    float* ysbuf   = (float*)(ws + carve((size_t)Bb * Kk * Ll * Dd * 4));
    float* y_comb  = (float*)(ws + carve((size_t)BL * Dd * 4));
    bf16*  yg_bf   = (bf16*)(ws + carve((size_t)BL * Dd * 2));

    // 1) LayerNorm -> bf16
    k_ln<<<BL, 256, 0, stream>>>(x, ln_g, ln_b, xn_bf);

    // 2) weight conversions
    {
        int n1 = E2 * Cc;  k_cvt<<<(n1 + 255) / 256, 256, 0, stream>>>(in_w, w_in_bf, n1);
        int n2 = Cc * Dd;  k_cvt<<<(n2 + 255) / 256, 256, 0, stream>>>(out_w, w_o_bf, n2);
        int n3 = Kk * CP * Dd;
        k_cvt_xproj<<<(n3 + 255) / 256, 256, 0, stream>>>(xproj_w, w_xp_bf);
    }

    // 3) input projection GEMM (WMMA): 288*48 tiles, 8 waves/block
    k_gemm_xz<<<(BL / 16) * (E2 / 16) / 8, 256, 0, stream>>>(xn_bf, w_in_bf, xz);

    // 4) depthwise conv + SiLU
    {
        size_t n = (size_t)BL * Dd;
        k_conv<<<(unsigned)((n + 255) / 256), 256, 0, stream>>>(xz, conv_w, conv_b, xc_f, xc_bf);
    }

    // 5) x_dbl GEMM (WMMA), weights TDM-staged in LDS, A rows permuted per direction
    k_gemm_xdbl<<<Bb * Kk * XDBL_TPB, 256, 0, stream>>>(xc_bf, w_xp_bf, xdbl);

    // 6) selective scan: one block per (b,k)
    k_scan<<<Bb * Kk, Dd, 0, stream>>>(xc_f, xdbl, dt_w, dt_b, A_logs, Ds, ysbuf);

    // 7) merge directions
    {
        size_t n = (size_t)BL * Dd;
        k_combine<<<(unsigned)((n + 255) / 256), 256, 0, stream>>>(ysbuf, y_comb);
    }

    // 8) output LN + gate
    k_ln_gate<<<BL, 512, 0, stream>>>(y_comb, xz, on_g, on_b, yg_bf);

    // 9) output GEMM + residual
    k_gemm_out<<<(BL / 16) * (Cc / 16) / 8, 256, 0, stream>>>(yg_bf, w_o_bf, x, out);
}
